// InwardSolve_26070451487158
// MI455X (gfx1250) — compile-verified, matrix operand order checked by reading, not run on hardware
//
#include <hip/hip_runtime.h>
#include <hip/hip_bf16.h>
#include <math.h>

// ---------------------------------------------------------------------------
// Types for CDNA5 WMMA (wave32): bf16 A/B fragments, f32 accumulator.
// ---------------------------------------------------------------------------
typedef __attribute__((ext_vector_type(16))) __bf16 v16bf;
typedef __attribute__((ext_vector_type(8)))  __bf16 v8bf;
typedef __attribute__((ext_vector_type(8)))  float  v8f;
typedef __attribute__((ext_vector_type(4)))  float  v4f;

__device__ __forceinline__ float gelu_f(float x) {
    // exact erf-based GELU (matches jax.nn.gelu(approximate=False))
    return 0.5f * x * (1.0f + erff(x * 0.70710678118654752440f));
}

// A fragment (16-bit A 16x32 layout): per lane, K-subset = two contiguous
// 8-float chunks -> 4x b128 loads + packed cvt, no scalar dword loads.
__device__ __forceinline__ v16bf load_frag_a(const float* __restrict__ arow, int kbase) {
    const v4f a0 = *(const v4f*)(arow + kbase);
    const v4f a1 = *(const v4f*)(arow + kbase + 4);
    const v4f a2 = *(const v4f*)(arow + kbase + 16);
    const v4f a3 = *(const v4f*)(arow + kbase + 20);
    v16bf v;
#pragma unroll
    for (int j = 0; j < 4; ++j) {
        v[j]      = (__bf16)a0[j];
        v[j + 4]  = (__bf16)a1[j];
        v[j + 8]  = (__bf16)a2[j];
        v[j + 12] = (__bf16)a3[j];
    }
    return v;
}

// B fragment from pre-transposed bf16 weights Wt[term][col][k]:
// two 16-byte loads, zero conversion VALU.
__device__ __forceinline__ v16bf load_frag_b(const __bf16* __restrict__ wcol, int kbase) {
    v8bf lo = *(const v8bf*)(wcol + kbase);
    v8bf hi = *(const v8bf*)(wcol + kbase + 16);
    return __builtin_shufflevector(lo, hi, 0,1,2,3,4,5,6,7,8,9,10,11,12,13,14,15);
}

// ---------------------------------------------------------------------------
// WMMA GEMM:  out[N,64] = sum_{k<kterms} A_k[N,64] @ W_k[64,64]  + bias
//             (+ optional addend[N,64]) (+ optional exact GELU)
// Wt = bf16, fragment-linear (col-major 64x64 per term). One wave per 16 rows.
// ---------------------------------------------------------------------------
__global__ void gemm_cheb_wmma(const float* __restrict__ A, size_t termStride, int kterms,
                               const __bf16* __restrict__ Wt, const float* __restrict__ bias,
                               const float* __restrict__ addend, float* __restrict__ out,
                               int Nn, int ntiles, int do_gelu) {
    int wid = blockIdx.x * (blockDim.x >> 5) + (threadIdx.x >> 5);
    if (wid >= ntiles) return;
    int lane = threadIdx.x & 31;
    int r0   = wid << 4;
    int lrow = lane & 15;
    int hi   = (lane >> 4) << 3;        // 0 or 8
    int arow = r0 + lrow; if (arow >= Nn) arow = Nn - 1;

    v8f acc[4] = {};
    for (int k = 0; k < kterms; ++k) {
        const float*  Ak = A  + (size_t)k * termStride + (size_t)arow * 64;
        const __bf16* Wk = Wt + (size_t)k * 64 * 64;
#pragma unroll
        for (int kh = 0; kh < 2; ++kh) {       // K = 0..31, 32..63
            int kbase = kh * 32 + hi;
            v16bf af = load_frag_a(Ak, kbase);
#pragma unroll
            for (int c = 0; c < 4; ++c) {      // 4 column tiles of 16
                v16bf bf = load_frag_b(Wk + (size_t)(c * 16 + lrow) * 64, kbase);
                acc[c] = __builtin_amdgcn_wmma_f32_16x16x32_bf16(
                    false, af, false, bf, (short)0, acc[c], false, false);
            }
        }
    }
    // C/D layout: element r of lane -> (M = r + hi, N = lane&15)
#pragma unroll
    for (int c = 0; c < 4; ++c) {
        int cc = c * 16 + lrow;
#pragma unroll
        for (int r = 0; r < 8; ++r) {
            int row = r0 + r + hi;
            if (row < Nn) {
                float v = acc[c][r] + bias[cc];
                if (addend) v += addend[(size_t)row * 64 + cc];
                if (do_gelu) v = gelu_f(v);
                out[(size_t)row * 64 + cc] = v;
            }
        }
    }
}

// Weight prep: Wt[t][col][k] = bf16(W[t][k][col])  (fragment-linear transpose)
__global__ void wprep_kernel(const float* __restrict__ W, __bf16* __restrict__ Wt, int terms) {
    int i = blockIdx.x * blockDim.x + threadIdx.x;
    if (i >= terms * 4096) return;
    int t = i >> 12, col = (i >> 6) & 63, k = i & 63;
    Wt[i] = (__bf16)W[t * 4096 + k * 64 + col];
}

// ---------------------------------------------------------------------------
// Elementwise kernels
// ---------------------------------------------------------------------------
__global__ void axb_kernel(const float* __restrict__ in, float a,
                           float* __restrict__ out, size_t n) {
    size_t i = (size_t)blockIdx.x * blockDim.x + threadIdx.x;
    if (i < n) out[i] = in ? a * in[i] : 0.0f;
}

__global__ void fill_i32(int* __restrict__ p, int v, int n) {
    int i = blockIdx.x * blockDim.x + threadIdx.x;
    if (i < n) p[i] = v;
}

__global__ void deg_kernel(const int* __restrict__ src, const float* __restrict__ ea,
                           float* __restrict__ deg, int E) {
    int e = blockIdx.x * blockDim.x + threadIdx.x;
    if (e < E) atomicAdd(&deg[src[e]], ea[e]);
}

__global__ void dis_kernel(const float* __restrict__ deg, float* __restrict__ dis, int Nn) {
    int i = blockIdx.x * blockDim.x + threadIdx.x;
    if (i < Nn) { float d = deg[i]; dis[i] = (d > 0.0f) ? rsqrtf(d) : 0.0f; }
}

__global__ void norm_kernel(const float* __restrict__ ea, const float* __restrict__ dis,
                            const int* __restrict__ src, const int* __restrict__ tgt,
                            float* __restrict__ nw, int E) {
    int e = blockIdx.x * blockDim.x + threadIdx.x;
    if (e < E) nw[e] = -ea[e] * dis[src[e]] * dis[tgt[e]];
}

// ---------------------------------------------------------------------------
// CSR build (counting sort by tgt): histogram -> 3-kernel scan -> placement
// ---------------------------------------------------------------------------
__global__ void hist_kernel(const int* __restrict__ tgt, int* __restrict__ cnt, int E) {
    int e = blockIdx.x * blockDim.x + threadIdx.x;
    if (e < E) atomicAdd(&cnt[tgt[e]], 1);
}

__global__ void blk_sum(const int* __restrict__ cnt, int* __restrict__ partial, int Nn) {
    __shared__ int sm[256];
    int i = blockIdx.x * 256 + threadIdx.x;
    sm[threadIdx.x] = (i < Nn) ? cnt[i] : 0;
    __syncthreads();
    for (int off = 128; off > 0; off >>= 1) {
        if ((int)threadIdx.x < off) sm[threadIdx.x] += sm[threadIdx.x + off];
        __syncthreads();
    }
    if (threadIdx.x == 0) partial[blockIdx.x] = sm[0];
}

__global__ void scan_partials(int* __restrict__ partial, int nb) {
    if (blockIdx.x == 0 && threadIdx.x == 0) {
        int run = 0;
        for (int i = 0; i < nb; ++i) { int v = partial[i]; partial[i] = run; run += v; }
    }
}

__global__ void blk_scan(const int* __restrict__ cnt, const int* __restrict__ partial,
                         int* __restrict__ rowptr, int Nn) {
    __shared__ int sm[256];
    int i = blockIdx.x * 256 + threadIdx.x;
    int v = (i < Nn) ? cnt[i] : 0;
    sm[threadIdx.x] = v;
    __syncthreads();
    for (int off = 1; off < 256; off <<= 1) {          // inclusive Hillis-Steele
        int t = ((int)threadIdx.x >= off) ? sm[threadIdx.x - off] : 0;
        __syncthreads();
        sm[threadIdx.x] += t;
        __syncthreads();
    }
    if (i < Nn) rowptr[i] = partial[blockIdx.x] + sm[threadIdx.x] - v;  // exclusive
}

__global__ void sort_scatter(const int* __restrict__ src, const int* __restrict__ tgt,
                             const float* __restrict__ nw, const int* __restrict__ rowptr,
                             int* __restrict__ fill, int* __restrict__ ssrc,
                             float* __restrict__ snrm, int E) {
    int e = blockIdx.x * blockDim.x + threadIdx.x;
    if (e >= E) return;
    int t = tgt[e];
    int pos = rowptr[t] + atomicAdd(&fill[t], 1);
    ssrc[pos] = src[e];
    snrm[pos] = nw[e];
}

// ---------------------------------------------------------------------------
// Atomic-free propagation (gather-reduce over CSR), with fused Chebyshev
// recurrence:  out[n] = scale * sum_{e in in(n)} nw[e]*hin[src[e]]  - sub[n]
// One wave per node; lane owns a float2 of the 64 features; (src, nw) pairs
// are loaded coalesced once per 32 edges and broadcast via shuffles.
// h (25.6 MB) is L2-resident on MI455X (192 MB L2), so gathers hit L2.
// ---------------------------------------------------------------------------
__global__ void prop_gather(const float* __restrict__ hin, const int* __restrict__ rowptr,
                            const int* __restrict__ ssrc, const float* __restrict__ snrm,
                            float scale, const float* __restrict__ sub,
                            float* __restrict__ outb, int Nn, int Ee) {
    int n = blockIdx.x * (blockDim.x >> 5) + (threadIdx.x >> 5);
    if (n >= Nn) return;
    int lane = threadIdx.x & 31;
    int beg = rowptr[n];
    int end = (n == Nn - 1) ? Ee : rowptr[n + 1];
    float a0 = 0.0f, a1 = 0.0f;
    for (int base = beg; base < end; base += 32) {
        int cnt = end - base; if (cnt > 32) cnt = 32;
        int idx = base + (lane < cnt ? lane : cnt - 1);
        int   es = ssrc[idx];
        float ew = snrm[idx];
        for (int j = 0; j < cnt; ++j) {
            int   s = __shfl(es, j, 32);
            float w = __shfl(ew, j, 32);
            float2 v = ((const float2*)(hin + (size_t)s * 64))[lane];
            a0 += w * v.x;
            a1 += w * v.y;
        }
    }
    float r0 = scale * a0, r1 = scale * a1;
    if (sub) {
        float2 sv = ((const float2*)(sub + (size_t)n * 64))[lane];
        r0 -= sv.x; r1 -= sv.y;
    }
    ((float2*)(outb + (size_t)n * 64))[lane] = make_float2(r0, r1);
}

// mid[n,d] = gelu( sum_{i<4} x[n,i]*w[i,d] + b[d] )
__global__ void embed1_kernel(const float* __restrict__ x, const float* __restrict__ w,
                              const float* __restrict__ b, float* __restrict__ mid, int Nn) {
    int i = blockIdx.x * blockDim.x + threadIdx.x;
    if (i >= Nn * 64) return;
    int n = i >> 6, d = i & 63;
    float s = b[d];
#pragma unroll
    for (int k = 0; k < 4; ++k) s += x[n * 4 + k] * w[k * 64 + d];
    mid[i] = gelu_f(s);
}

// out[n] = gelu(dot(h[n], d1_w)+d1_b)*d2_w + d2_b
__global__ void down_kernel(const float* __restrict__ h, const float* __restrict__ w1,
                            const float* __restrict__ b1, const float* __restrict__ w2,
                            const float* __restrict__ b2, float* __restrict__ out, int Nn) {
    int n = blockIdx.x * blockDim.x + threadIdx.x;
    if (n >= Nn) return;
    const float* hr = h + (size_t)n * 64;
    float s = b1[0];
#pragma unroll
    for (int j = 0; j < 64; ++j) s += hr[j] * w1[j];
    out[n] = gelu_f(s) * w2[0] + b2[0];
}

// ---------------------------------------------------------------------------
// Host-side orchestration
// ---------------------------------------------------------------------------
static inline int cdiv(long a, long b) { return (int)((a + b - 1) / b); }

extern "C" void kernel_launch(void* const* d_in, const int* in_sizes, int n_in,
                              void* d_out, int out_size, void* d_ws, size_t ws_size,
                              hipStream_t stream) {
    const float* x    = (const float*)d_in[0];
    const int*   ei   = (const int*)  d_in[1];
    const float* ea   = (const float*)d_in[2];
    const float* e1_w = (const float*)d_in[3];
    const float* e1_b = (const float*)d_in[4];
    const float* e2_w = (const float*)d_in[5];
    const float* e2_b = (const float*)d_in[6];
    const float* cw1  = (const float*)d_in[7];
    const float* cb1  = (const float*)d_in[8];
    const float* cw2  = (const float*)d_in[9];
    const float* cb2  = (const float*)d_in[10];
    const float* lw   = (const float*)d_in[11];
    const float* lb   = (const float*)d_in[12];
    const float* d1_w = (const float*)d_in[13];
    const float* d1_b = (const float*)d_in[14];
    const float* d2_w = (const float*)d_in[15];
    const float* d2_b = (const float*)d_in[16];
    float* out = (float*)d_out;

    const int Nn = in_sizes[0] / 4;          // 100000
    const int Ee = in_sizes[2];              // 1600000
    const int Kc = 5;                        // Chebyshev order
    const int* src = ei;
    const int* tgt = ei + Ee;
    const size_t nd = (size_t)Nn * 64;
    const int nb = cdiv(Nn, 256);            // scan blocks

    // workspace carve (16B-aligned bump allocator)
    char* p = (char*)d_ws;
    auto alloc = [&](size_t bytes) { char* r = p; p += (bytes + 15) & ~((size_t)15); return r; };
    float* deg    = (float*)alloc((size_t)Nn * 4);
    float* dis    = (float*)alloc((size_t)Nn * 4);
    float* nw     = (float*)alloc((size_t)Ee * 4);
    float* snrm   = (float*)alloc((size_t)Ee * 4);
    float* h      = (float*)alloc(nd * 4);
    float* t      = (float*)alloc(nd * 4);
    float* f      = (float*)alloc(nd * 4);
    float* Tx     = (float*)alloc(nd * 4 * Kc);
    int*   cnt    = (int*)  alloc((size_t)Nn * 4);   // histogram, then fill counters
    int*   rowptr = (int*)  alloc((size_t)Nn * 4);
    int*   ssrc   = (int*)  alloc((size_t)Ee * 4);
    int*   part   = (int*)  alloc((size_t)nb * 4);
    __bf16* wtE2  = (__bf16*)alloc((size_t)1  * 4096 * 2);
    __bf16* wtC1  = (__bf16*)alloc((size_t)15 * 4096 * 2);
    __bf16* wtC2  = (__bf16*)alloc((size_t)15 * 4096 * 2);
    __bf16* wtL   = (__bf16*)alloc((size_t)3  * 4096 * 2);

    const int ntiles  = (Nn + 15) / 16;
    const int gBlocks = cdiv(ntiles, 8);     // 8 waves (row tiles) per 256-thr block
    const int nWaves  = cdiv(Nn, 8);         // 8 node-waves per block (prop)

    // 0) weight prep: bf16, fragment-linear transpose (tiny, L2-resident)
    wprep_kernel<<<cdiv(1  * 4096, 256), 256, 0, stream>>>(e2_w, wtE2, 1);
    wprep_kernel<<<cdiv(15 * 4096, 256), 256, 0, stream>>>(cw1, wtC1, 15);
    wprep_kernel<<<cdiv(15 * 4096, 256), 256, 0, stream>>>(cw2, wtC2, 15);
    wprep_kernel<<<cdiv(3  * 4096, 256), 256, 0, stream>>>(lw,  wtL,  3);

    // 1) normalization weights
    axb_kernel<<<cdiv(Nn, 256), 256, 0, stream>>>(nullptr, 0.0f, deg, (size_t)Nn);
    deg_kernel<<<cdiv(Ee, 256), 256, 0, stream>>>(src, ea, deg, Ee);
    dis_kernel<<<cdiv(Nn, 256), 256, 0, stream>>>(deg, dis, Nn);
    norm_kernel<<<cdiv(Ee, 256), 256, 0, stream>>>(ea, dis, src, tgt, nw, Ee);

    // 2) CSR build by tgt (once; reused by all 24 propagations)
    fill_i32<<<cdiv(Nn, 256), 256, 0, stream>>>(cnt, 0, Nn);
    hist_kernel<<<cdiv(Ee, 256), 256, 0, stream>>>(tgt, cnt, Ee);
    blk_sum<<<nb, 256, 0, stream>>>(cnt, part, Nn);
    scan_partials<<<1, 32, 0, stream>>>(part, nb);
    blk_scan<<<nb, 256, 0, stream>>>(cnt, part, rowptr, Nn);
    fill_i32<<<cdiv(Nn, 256), 256, 0, stream>>>(cnt, 0, Nn);
    sort_scatter<<<cdiv(Ee, 256), 256, 0, stream>>>(src, tgt, nw, rowptr, cnt, ssrc, snrm, Ee);

    // 3) encoder MLP: h = gelu(gelu(x@e1_w+e1_b)@e2_w+e2_b)
    embed1_kernel<<<cdiv((long)Nn * 64, 256), 256, 0, stream>>>(x, e1_w, e1_b, t, Nn);
    gemm_cheb_wmma<<<gBlocks, 256, 0, stream>>>(t, nd, 1, wtE2, e2_b, nullptr, h, Nn, ntiles, 1);

    // Chebyshev basis: Tx0=X; Tx1=prop(X); Txk = 2*prop(Tx_{k-1}) - Tx_{k-2}
    // (recurrence fused into the gather epilogue; no atomics, no init passes)
    auto build_cheb = [&](const float* X) {
        axb_kernel<<<cdiv((long)nd, 256), 256, 0, stream>>>(X, 1.0f, Tx, nd);
        prop_gather<<<nWaves, 256, 0, stream>>>(Tx, rowptr, ssrc, snrm, 1.0f, nullptr,
                                                Tx + nd, Nn, Ee);
        for (int k = 2; k < Kc; ++k) {
            prop_gather<<<nWaves, 256, 0, stream>>>(Tx + (size_t)(k - 1) * nd, rowptr, ssrc, snrm,
                                                    2.0f, Tx + (size_t)(k - 2) * nd,
                                                    Tx + (size_t)k * nd, Nn, Ee);
        }
    };

    // 4) three residual Chebyshev blocks
    for (int b = 0; b < 3; ++b) {
        const __bf16* wc1 = wtC1 + (size_t)b * Kc * 4096;
        const __bf16* wc2 = wtC2 + (size_t)b * Kc * 4096;
        const __bf16* wlb = wtL  + (size_t)b * 4096;
        const float* cb1b = cb1 + (size_t)b * 64;
        const float* cb2b = cb2 + (size_t)b * 64;
        const float* lbb  = lb  + (size_t)b * 64;

        build_cheb(h);
        gemm_cheb_wmma<<<gBlocks, 256, 0, stream>>>(Tx, nd, Kc, wc1, cb1b, nullptr, t, Nn, ntiles, 1);
        build_cheb(t);
        gemm_cheb_wmma<<<gBlocks, 256, 0, stream>>>(Tx, nd, Kc, wc2, cb2b, nullptr, f, Nn, ntiles, 0);
        // h = gelu(f + h@lw + lb)  (in-place safe: each wave reads its rows first)
        gemm_cheb_wmma<<<gBlocks, 256, 0, stream>>>(h, nd, 1, wlb, lbb, f, h, Nn, ntiles, 1);
    }

    // 5) decoder
    down_kernel<<<cdiv(Nn, 256), 256, 0, stream>>>(h, d1_w, d1_b, d2_w, d2_b, out, Nn);
}
